// SparseGATLayer_76141180223722
// MI455X (gfx1250) — compile-verified, matrix operand order checked by reading, not run on hardware
//
#include <hip/hip_runtime.h>
#include <hip/hip_bf16.h>
#include <math.h>

#define N_NODES 100000
#define N_EDGES 3200000
#define IN_DIM  256
#define OUT_DIM 128
#define ALPHA   0.1f
#define GAT_EPS 9e-15f

typedef __attribute__((ext_vector_type(16))) _Float16 v16h;
typedef __attribute__((ext_vector_type(8)))  _Float16 v8h;
typedef __attribute__((ext_vector_type(8)))  float    v8f;

// ---------------------------------------------------------------------------
// 0) init: zero output accumulator + rowsum, gmax = -inf (runs every call)
// ---------------------------------------------------------------------------
__global__ void gat_init(float* __restrict__ out, float* __restrict__ rowsum,
                         unsigned int* __restrict__ gmax) {
    long long i = (long long)blockIdx.x * blockDim.x + threadIdx.x;
    long long stride = (long long)gridDim.x * blockDim.x;
    const long long tot = (long long)N_NODES * OUT_DIM;
    for (long long k = i; k < tot; k += stride) out[k] = 0.0f;
    for (long long k = i; k < N_NODES; k += stride) rowsum[k] = 0.0f;
    if (i == 0) *gmax = 0xFF800000u; // -inf bits
}

// ---------------------------------------------------------------------------
// helpers for the WMMA GEMM
// ---------------------------------------------------------------------------
__device__ __forceinline__ v16h gat_load_a(const float* __restrict__ xr,
                                           int kk, int koff) {
    // A fragment 16x32 f16: halves 0-7 = K[kk+koff .. +7],
    //                       halves 8-15 = K[kk+16+koff .. +7]
    float4 f0 = *(const float4*)(xr + kk + koff);
    float4 f1 = *(const float4*)(xr + kk + koff + 4);
    float4 f2 = *(const float4*)(xr + kk + 16 + koff);
    float4 f3 = *(const float4*)(xr + kk + 16 + koff + 4);
    v16h a;
    a[0]=(_Float16)f0.x;  a[1]=(_Float16)f0.y;  a[2]=(_Float16)f0.z;  a[3]=(_Float16)f0.w;
    a[4]=(_Float16)f1.x;  a[5]=(_Float16)f1.y;  a[6]=(_Float16)f1.z;  a[7]=(_Float16)f1.w;
    a[8]=(_Float16)f2.x;  a[9]=(_Float16)f2.y;  a[10]=(_Float16)f2.z; a[11]=(_Float16)f2.w;
    a[12]=(_Float16)f3.x; a[13]=(_Float16)f3.y; a[14]=(_Float16)f3.z; a[15]=(_Float16)f3.w;
    return a;
}

__device__ __forceinline__ void gat_load_b(const _Float16* __restrict__ Wt,
                                           int kk, int koff, int lrow,
                                           v16h* __restrict__ b) {
    #pragma unroll
    for (int t = 0; t < 8; ++t) {
        const _Float16* wb = &Wt[(t * 16 + lrow) * IN_DIM + kk + koff];
        v8h blo = *(const v8h*)wb;            // 16B LDS load, K contiguous
        v8h bhi = *(const v8h*)(wb + 16);     // K + 16
        b[t] = __builtin_shufflevector(blo, bhi,
                    0,1,2,3,4,5,6,7,8,9,10,11,12,13,14,15);
    }
}

// ---------------------------------------------------------------------------
// 1) h = x @ W   (f16 WMMA, fp32 accumulate). W transposed into LDS (64 KB).
//    Block = 256 threads = 8 waves; block computes 128 rows x 128 cols.
//    Each wave: 16 rows, 8 tiles of 16x16, K-loop of 8 steps of 32.
//    A and B fragments are double-buffered: all LDS/global loads for step
//    kk+32 issue before the 8 WMMAs of step kk, so the matrix pipe runs
//    back-to-back and no WMMA->VALU hazard NOPs are needed.
// ---------------------------------------------------------------------------
__global__ __launch_bounds__(256)
__attribute__((amdgpu_waves_per_eu(2)))
void gat_gemm(const float* __restrict__ x,
              const float* __restrict__ W,
              float* __restrict__ h) {
    __shared__ _Float16 Wt[OUT_DIM * IN_DIM]; // Wt[n][k], 64 KB

    const int tid = threadIdx.x;
    for (int i = tid; i < IN_DIM * OUT_DIM; i += 256) {
        int k = i >> 7;       // W is [256][128] row-major
        int n = i & 127;
        Wt[n * IN_DIM + k] = (_Float16)W[i];
    }
    __syncthreads();

    const int wave = tid >> 5;
    const int lane = tid & 31;
    const int koff = (lane >> 4) << 3;   // 0 for lanes 0-15, 8 for lanes 16-31
    const int lrow = lane & 15;
    const int rowBase = blockIdx.x * 128 + wave * 16;

    int arow = rowBase + lrow;
    if (arow >= N_NODES) arow = N_NODES - 1;      // clamp: keep EXEC all-ones
    const float* __restrict__ xr = x + (long long)arow * IN_DIM;

    v8f acc[8] = {};

    v16h bcur[8], bnxt[8];
    gat_load_b(Wt, 0, koff, lrow, bcur);
    v16h a = gat_load_a(xr, 0, koff);

    #pragma unroll
    for (int kk = 0; kk < IN_DIM; kk += 32) {
        v16h anxt;
        if (kk + 32 < IN_DIM) {
            __builtin_prefetch(xr + kk + 64 + koff, 0, 1);
            gat_load_b(Wt, kk + 32, koff, lrow, bnxt);   // next-step LDS clause
            anxt = gat_load_a(xr, kk + 32, koff);        // next-step A (fresh regs)
        }
        #pragma unroll
        for (int t = 0; t < 8; ++t) {
            acc[t] = __builtin_amdgcn_wmma_f32_16x16x32_f16(
                        false, a, false, bcur[t], (short)0, acc[t], false, false);
        }
        if (kk + 32 < IN_DIM) {
            a = anxt;
            #pragma unroll
            for (int t = 0; t < 8; ++t) bcur[t] = bnxt[t];  // renamed away
        }
    }

    // C/D layout: lane L, VGPR r -> M = r + (L>=16 ? 8 : 0), N = L&15
    #pragma unroll
    for (int r = 0; r < 8; ++r) {
        const int row = rowBase + koff + r;
        if (row < N_NODES) {
            float* __restrict__ hr = h + (long long)row * OUT_DIM + lrow;
            #pragma unroll
            for (int t = 0; t < 8; ++t)
                hr[t * 16] = acc[t][r];
        }
    }
}

// ---------------------------------------------------------------------------
// 2) score_l / score_r : one wave per row, lane handles 4 contiguous dims
// ---------------------------------------------------------------------------
__global__ __launch_bounds__(256) void gat_scores(const float* __restrict__ h,
                                                  const float* __restrict__ attn,
                                                  float* __restrict__ sl,
                                                  float* __restrict__ sr) {
    const int wave = threadIdx.x >> 5;
    const int lane = threadIdx.x & 31;
    const int row  = blockIdx.x * 8 + wave;
    if (row >= N_NODES) return;

    float4 hv = *(const float4*)(h + (long long)row * OUT_DIM + lane * 4);
    float4 al = *(const float4*)(attn + lane * 4);
    float4 ar = *(const float4*)(attn + OUT_DIM + lane * 4);
    float dl = hv.x*al.x + hv.y*al.y + hv.z*al.z + hv.w*al.w;
    float dr = hv.x*ar.x + hv.y*ar.y + hv.z*ar.z + hv.w*ar.w;
    #pragma unroll
    for (int off = 16; off > 0; off >>= 1) {
        dl += __shfl_down(dl, off, 32);
        dr += __shfl_down(dr, off, 32);
    }
    if (lane == 0) { sl[row] = dl; sr[row] = dr; }
}

// ---------------------------------------------------------------------------
// 3) per-edge leaky-relu score + global max
// ---------------------------------------------------------------------------
__device__ __forceinline__ void atomicMaxF(float* addr, float v) {
    if (v >= 0.0f) atomicMax((int*)addr, __float_as_int(v));
    else           atomicMin((unsigned int*)addr, __float_as_uint(v));
}

__global__ __launch_bounds__(256) void gat_edge_score(const int* __restrict__ src,
                                                      const int* __restrict__ dst,
                                                      const float* __restrict__ sl,
                                                      const float* __restrict__ sr,
                                                      float* __restrict__ ea,
                                                      float* __restrict__ gmax) {
    __shared__ float red[256];
    float lmax = -INFINITY;
    for (int e = blockIdx.x * blockDim.x + threadIdx.x; e < N_EDGES;
         e += gridDim.x * blockDim.x) {
        float v = sl[src[e]] + sr[dst[e]];
        v = (v > 0.0f) ? v : ALPHA * v;
        ea[e] = v;
        lmax = fmaxf(lmax, v);
    }
    red[threadIdx.x] = lmax;
    __syncthreads();
    for (int s = 128; s > 0; s >>= 1) {
        if (threadIdx.x < s)
            red[threadIdx.x] = fmaxf(red[threadIdx.x], red[threadIdx.x + s]);
        __syncthreads();
    }
    if (threadIdx.x == 0) atomicMaxF(gmax, red[0]);
}

// ---------------------------------------------------------------------------
// 4) scatter-add: one wave per edge; lane covers 4 of the 128 output dims.
//    h and out both fit in the 192 MB L2, so f32 atomics stay on-chip.
// ---------------------------------------------------------------------------
__global__ __launch_bounds__(256) void gat_aggregate(const int* __restrict__ src,
                                                     const int* __restrict__ dst,
                                                     const float* __restrict__ ea,
                                                     const float* __restrict__ h,
                                                     const float* __restrict__ gmaxp,
                                                     float* __restrict__ out,
                                                     float* __restrict__ rowsum) {
    const int lane   = threadIdx.x & 31;
    const int wave   = (blockIdx.x * blockDim.x + threadIdx.x) >> 5;
    const int nwaves = (gridDim.x * blockDim.x) >> 5;
    const float gmax = *gmaxp;

    for (int e = wave; e < N_EDGES; e += nwaves) {
        const int s = src[e];
        const int d = dst[e];
        const float w = __expf(ea[e] - gmax);
        float4 g = *(const float4*)(h + (long long)d * OUT_DIM + lane * 4);
        float* o = out + (long long)s * OUT_DIM + lane * 4;
        atomicAdd(o + 0, w * g.x);
        atomicAdd(o + 1, w * g.y);
        atomicAdd(o + 2, w * g.z);
        atomicAdd(o + 3, w * g.w);
        if (lane == 0) atomicAdd(rowsum + s, w);
    }
}

// ---------------------------------------------------------------------------
// 5) finalize: out = leaky_relu(out / (rowsum + eps))
// ---------------------------------------------------------------------------
__global__ void gat_final(float* __restrict__ out,
                          const float* __restrict__ rowsum) {
    long long stride = (long long)gridDim.x * blockDim.x;
    const long long tot = (long long)N_NODES * OUT_DIM;
    for (long long i = (long long)blockIdx.x * blockDim.x + threadIdx.x;
         i < tot; i += stride) {
        float v = out[i] / (rowsum[i >> 7] + GAT_EPS);
        out[i] = (v > 0.0f) ? v : ALPHA * v;
    }
}

// ---------------------------------------------------------------------------
extern "C" void kernel_launch(void* const* d_in, const int* in_sizes, int n_in,
                              void* d_out, int out_size, void* d_ws, size_t ws_size,
                              hipStream_t stream) {
    const float* x    = (const float*)d_in[0];
    const int*   ei   = (const int*)d_in[1];
    const float* W    = (const float*)d_in[2];
    const float* attn = (const float*)d_in[3];
    const int* src = ei;             // edge_index[0]
    const int* dst = ei + N_EDGES;   // edge_index[1]

    float* ws     = (float*)d_ws;
    float* h      = ws;                               // 12,800,000
    float* sl     = h  + (size_t)N_NODES * OUT_DIM;   //    100,000
    float* sr     = sl + N_NODES;                     //    100,000
    float* ea     = sr + N_NODES;                     //  3,200,000
    float* rowsum = ea + N_EDGES;                     //    100,000
    float* gmax   = rowsum + N_NODES;                 //          1
    float* out    = (float*)d_out;

    gat_init<<<1024, 256, 0, stream>>>(out, rowsum, (unsigned int*)gmax);
    gat_gemm<<<(N_NODES + 127) / 128, 256, 0, stream>>>(x, W, h);
    gat_scores<<<(N_NODES + 7) / 8, 256, 0, stream>>>(h, attn, sl, sr);
    gat_edge_score<<<2048, 256, 0, stream>>>(src, dst, sl, sr, ea, gmax);
    gat_aggregate<<<8192, 256, 0, stream>>>(src, dst, ea, h, gmax, out, rowsum);
    gat_final<<<4096, 256, 0, stream>>>(out, rowsum);
}
